// PIN_5944234738100
// MI455X (gfx1250) — compile-verified, matrix operand order checked by reading, not run on hardware
//
#include <hip/hip_runtime.h>
#include <math.h>

#define CONT_F 13
#define CATE_F 26
#define FF     39
#define D      40
#define COMB   741
#define BATCH  2048
#define SUB0   40
#define SUB1   5
#define HH     100
#define FIRST_IN 5265          // COMB*SUB1 + FF*D
#define KX     5280            // FIRST_IN padded to multiple of 32 (165*32)
#define NPAD   112             // HH padded to multiple of 16
#define KSTEPS (KX / 32)       // 165
#define NT_FC0 7               // NPAD / 16

typedef __attribute__((ext_vector_type(16))) __bf16 v16bf;
typedef __attribute__((ext_vector_type(8)))  float  v8f;

__device__ __forceinline__ unsigned pack2bf(float a, float b) {
    union { __bf16 h[2]; unsigned u; } p;
    p.h[0] = (__bf16)a; p.h[1] = (__bf16)b;
    return p.u;
}

// ---------------------------------------------------------------------------
// Kernel 1: build bf16 feature matrix X[:, 0:1560] = e, zero K-padding tail
// ---------------------------------------------------------------------------
__global__ void k_embed(const float* __restrict__ conts,
                        const int*   __restrict__ cates,
                        const float* __restrict__ emb,
                        __bf16* __restrict__ Xb) {
    int b = blockIdx.x;
    for (int idx = threadIdx.x; idx < FF * D; idx += blockDim.x) {
        int f = idx / D, d = idx % D;
        float v;
        if (f < CONT_F) v = emb[f * D + d] * conts[b * CONT_F + f];
        else            v = emb[(size_t)cates[b * CATE_F + (f - CONT_F)] * D + d];
        Xb[(size_t)b * KX + idx] = (__bf16)v;
    }
    for (int idx = threadIdx.x; idx < KX - FIRST_IN; idx += blockDim.x)
        Xb[(size_t)b * KX + FIRST_IN + idx] = (__bf16)0.f;
}

// ---------------------------------------------------------------------------
// Kernel 2: fc0 weights f32 (5265x100) -> bf16 packed in WMMA B-fragment
// order: u32[ks][nt][lane][8], two bf16 per dword. Total KSTEPS*7*32*8 dwords.
// ---------------------------------------------------------------------------
__global__ void k_w0cvt(const float* __restrict__ w, unsigned* __restrict__ wb) {
    int idx = blockIdx.x * blockDim.x + threadIdx.x;
    if (idx >= KSTEPS * NT_FC0 * 32 * 8) return;
    int j    = idx & 7;
    int lane = (idx >> 3) & 31;
    int g    = idx >> 8;            // ks*7 + nt
    int nt   = g % NT_FC0;
    int ks   = g / NT_FC0;
    int half = lane >> 4, l16 = lane & 15;
    int n    = nt * 16 + l16;
    int kb   = half ? 8 : 0;
    int kl   = (j < 4) ? (kb + 2 * j) : (kb + 16 + 2 * (j - 4));
    int k0   = ks * 32 + kl;
    float a  = (k0     < FIRST_IN && n < HH) ? w[(size_t)k0 * HH + n]       : 0.f;
    float b2 = (k0 + 1 < FIRST_IN && n < HH) ? w[(size_t)(k0 + 1) * HH + n] : 0.f;
    wb[idx] = pack2bf(a, b2);
}

// ---------------------------------------------------------------------------
// Kernel 3: per-comb sub-network. grid (741, 32), block 128 (4 waves).
// z(64x120) @ w0(120x40) via bf16 WMMA, LN+ReLU, @ w1(40x5), LN+ReLU.
// B operand pre-packed in fragment layout in LDS -> contiguous b128 reads.
// ---------------------------------------------------------------------------
__global__ void __launch_bounds__(128)
k_comb(__bf16* __restrict__ Xb,
       const float* __restrict__ sub_w0, const float* __restrict__ sub_b0,
       const float* __restrict__ sln0_w, const float* __restrict__ sln0_b,
       const float* __restrict__ sub_w1, const float* __restrict__ sub_b1,
       const float* __restrict__ sln1_w, const float* __restrict__ sln1_b) {
    __shared__ unsigned sWp[4 * 3 * 32 * 8]; // packed B frags [ks][nt][lane][8]
    __shared__ __bf16   sZ[64 * 128];        // z [row][k]
    __shared__ float    sH[64 * 48];         // first-layer result

    int comb = blockIdx.x;
    int r = 0, rem = comb;
    while (rem >= (FF - 1 - r)) { rem -= (FF - 1 - r); ++r; }
    int cpos = r + 1 + rem;
    int b0 = blockIdx.y * 64;

    // stage w0 packed into B-fragment layout
    const float* w0g = sub_w0 + (size_t)comb * (3 * D * SUB0);
    for (int idx = threadIdx.x; idx < 4 * 3 * 256; idx += 128) {
        int j    = idx & 7;
        int lane = (idx >> 3) & 31;
        int g    = idx >> 8;          // ks*3 + nt
        int nt   = g % 3;
        int ks   = g / 3;
        int half = lane >> 4, l16 = lane & 15;
        int n    = nt * 16 + l16;
        int kb   = half ? 8 : 0;
        int kl   = (j < 4) ? (kb + 2 * j) : (kb + 16 + 2 * (j - 4));
        int k0   = ks * 32 + kl;
        float a  = (k0     < 3 * D && n < SUB0) ? w0g[k0 * SUB0 + n]       : 0.f;
        float b2 = (k0 + 1 < 3 * D && n < SUB0) ? w0g[(k0 + 1) * SUB0 + n] : 0.f;
        sWp[idx] = pack2bf(a, b2);
    }
    // stage z = [p, q, p*q]
    for (int idx = threadIdx.x; idx < 64 * 128; idx += 128) {
        int row = idx >> 7, k = idx & 127;
        const __bf16* xr = Xb + (size_t)(b0 + row) * KX;
        float v;
        if (k < D)            v = (float)xr[r * D + k];
        else if (k < 2 * D)   v = (float)xr[cpos * D + (k - D)];
        else if (k < 3 * D) { int d = k - 2 * D;
                              v = (float)xr[r * D + d] * (float)xr[cpos * D + d]; }
        else                  v = 0.f;
        sZ[idx] = (__bf16)v;
    }
    __syncthreads();

    int lane = threadIdx.x & 31, wave = threadIdx.x >> 5;
    int half = lane >> 4, l16 = lane & 15;

    // cache all 4 A fragments (16 rows x 128 K) in registers
    union fragU { v16bf v; unsigned u[8]; };
    fragU Af[4];
#pragma unroll
    for (int ks = 0; ks < 4; ++ks) {
        const unsigned* zrow =
            (const unsigned*)(sZ + (wave * 16 + l16) * 128) + ks * 16 + half * 4;
        Af[ks].u[0] = zrow[0];  Af[ks].u[1] = zrow[1];
        Af[ks].u[2] = zrow[2];  Af[ks].u[3] = zrow[3];
        Af[ks].u[4] = zrow[8];  Af[ks].u[5] = zrow[9];
        Af[ks].u[6] = zrow[10]; Af[ks].u[7] = zrow[11];
    }

#pragma unroll
    for (int nt = 0; nt < 3; ++nt) {
        v8f acc = {};
#pragma unroll
        for (int ks = 0; ks < 4; ++ks) {
            fragU Bf;
            const unsigned* bp = sWp + ((ks * 3 + nt) * 32 + lane) * 8;
#pragma unroll
            for (int j = 0; j < 8; ++j) Bf.u[j] = bp[j];
            acc = __builtin_amdgcn_wmma_f32_16x16x32_bf16(
                false, Af[ks].v, false, Bf.v, (short)0, acc, false, false);
        }
        int mrow = wave * 16 + (half ? 8 : 0);
#pragma unroll
        for (int v = 0; v < 8; ++v)
            sH[(mrow + v) * 48 + nt * 16 + l16] = acc[v];
    }
    __syncthreads();

    if (threadIdx.x < 64) {
        int row = threadIdx.x, b = b0 + row;
        float a0[SUB0];
        float mean = 0.f;
        for (int i = 0; i < SUB0; ++i) {
            float x = sH[row * 48 + i] + sub_b0[comb * SUB0 + i];
            a0[i] = x; mean += x;
        }
        mean *= (1.f / SUB0);
        float var = 0.f;
        for (int i = 0; i < SUB0; ++i) { float d = a0[i] - mean; var += d * d; }
        var *= (1.f / SUB0);
        float rs = rsqrtf(var + 1e-5f);
        for (int i = 0; i < SUB0; ++i) {
            float y = (a0[i] - mean) * rs * sln0_w[i] + sln0_b[i];
            a0[i] = y > 0.f ? y : 0.f;
        }
        float h1[SUB1]; float m1 = 0.f;
        for (int j = 0; j < SUB1; ++j) {
            float s = sub_b1[comb * SUB1 + j];
            for (int i = 0; i < SUB0; ++i)
                s += a0[i] * sub_w1[(size_t)comb * (SUB0 * SUB1) + i * SUB1 + j];
            h1[j] = s; m1 += s;
        }
        m1 *= (1.f / SUB1);
        float v1 = 0.f;
        for (int j = 0; j < SUB1; ++j) { float d = h1[j] - m1; v1 += d * d; }
        v1 *= (1.f / SUB1);
        float rs1 = rsqrtf(v1 + 1e-5f);
        __bf16* xo = Xb + (size_t)b * KX + FF * D + comb * SUB1;
        for (int j = 0; j < SUB1; ++j) {
            float y = (h1[j] - m1) * rs1 * sln1_w[j] + sln1_b[j];
            xo[j] = (__bf16)(y > 0.f ? y : 0.f);
        }
    }
}

// ---------------------------------------------------------------------------
// Kernel 4: fc0 GEMM (2048x5280 bf16) @ packed (5280x112 bf16) + LN + ReLU.
// grid 32 blocks x 128 threads; each wave: 16 rows x 112 cols, K-loop 165.
// B staged as packed fragments (coalesced dword copy, contiguous b128 reads).
// ---------------------------------------------------------------------------
__global__ void __launch_bounds__(128)
k_fc0(const __bf16* __restrict__ Xb, const unsigned* __restrict__ Wbp,
      const float* __restrict__ fc0_b, const float* __restrict__ ln0_w,
      const float* __restrict__ ln0_b, float* __restrict__ h0) {
    __shared__ __bf16   sA[64 * 32];
    __shared__ unsigned sBp[NT_FC0 * 32 * 8];   // 1792 dwords
    __shared__ float    sH[64 * NPAD];

    int b0 = blockIdx.x * 64;
    int lane = threadIdx.x & 31, wave = threadIdx.x >> 5;
    int half = lane >> 4, l16 = lane & 15;

    v8f zero = {};
    v8f acc[NT_FC0];
#pragma unroll
    for (int t = 0; t < NT_FC0; ++t) acc[t] = zero;

    union fragU { v16bf v; unsigned u[8]; };

    for (int ks = 0; ks < KSTEPS; ++ks) {
        __syncthreads();
        for (int idx = threadIdx.x; idx < 64 * 32; idx += 128) {
            int row = idx >> 5, kk = idx & 31;
            sA[idx] = Xb[(size_t)(b0 + row) * KX + ks * 32 + kk];
        }
        const unsigned* wsrc = Wbp + (size_t)ks * (NT_FC0 * 256);
        for (int idx = threadIdx.x; idx < NT_FC0 * 256; idx += 128)
            sBp[idx] = wsrc[idx];
        __syncthreads();

        fragU Af;
        const unsigned* ar = (const unsigned*)(sA + (wave * 16 + l16) * 32) + half * 4;
        Af.u[0] = ar[0];  Af.u[1] = ar[1];  Af.u[2] = ar[2];  Af.u[3] = ar[3];
        Af.u[4] = ar[8];  Af.u[5] = ar[9];  Af.u[6] = ar[10]; Af.u[7] = ar[11];

#pragma unroll
        for (int nt = 0; nt < NT_FC0; ++nt) {
            fragU Bf;
            const unsigned* bp = sBp + (nt * 32 + lane) * 8;
#pragma unroll
            for (int j = 0; j < 8; ++j) Bf.u[j] = bp[j];
            acc[nt] = __builtin_amdgcn_wmma_f32_16x16x32_bf16(
                false, Af.v, false, Bf.v, (short)0, acc[nt], false, false);
        }
    }

    int mrow = wave * 16 + (half ? 8 : 0);
#pragma unroll
    for (int nt = 0; nt < NT_FC0; ++nt)
#pragma unroll
        for (int v = 0; v < 8; ++v)
            sH[(mrow + v) * NPAD + nt * 16 + l16] = acc[nt][v];
    __syncthreads();

    if (threadIdx.x < 64) {
        int row = threadIdx.x, b = b0 + row;
        float sum = 0.f, sq = 0.f;
        for (int i = 0; i < HH; ++i) {
            float x = sH[row * NPAD + i] + fc0_b[i];
            sum += x; sq += x * x;
        }
        float mean = sum * (1.f / HH);
        float var  = sq * (1.f / HH) - mean * mean;
        float rs   = rsqrtf(var + 1e-5f);
        for (int i = 0; i < HH; ++i) {
            float x = sH[row * NPAD + i] + fc0_b[i];
            float y = (x - mean) * rs * ln0_w[i] + ln0_b[i];
            h0[(size_t)b * HH + i] = y > 0.f ? y : 0.f;
        }
    }
}

// ---------------------------------------------------------------------------
// Kernel 5: fc1 + LN + ReLU + out + sigmoid. One thread per batch row.
// ---------------------------------------------------------------------------
__global__ void __launch_bounds__(64)
k_head(const float* __restrict__ h0, const float* __restrict__ fc1_w,
       const float* __restrict__ fc1_b, const float* __restrict__ ln1_w,
       const float* __restrict__ ln1_b, const float* __restrict__ out_w,
       const float* __restrict__ out_b, float* __restrict__ out) {
    __shared__ float sX[64 * HH];
    __shared__ float sY[64 * HH];
    int t = threadIdx.x;
    int b = blockIdx.x * 64 + t;
    for (int i = 0; i < HH; ++i) sX[t * HH + i] = h0[(size_t)b * HH + i];
    float sum = 0.f, sq = 0.f;
    for (int j = 0; j < HH; ++j) {
        float s = fc1_b[j];
        for (int i = 0; i < HH; ++i) s += sX[t * HH + i] * fc1_w[i * HH + j];
        sY[t * HH + j] = s;
        sum += s; sq += s * s;
    }
    float mean = sum * (1.f / HH);
    float var  = sq * (1.f / HH) - mean * mean;
    float rs   = rsqrtf(var + 1e-5f);
    float o = out_b[0];
    for (int j = 0; j < HH; ++j) {
        float y = (sY[t * HH + j] - mean) * rs * ln1_w[j] + ln1_b[j];
        y = y > 0.f ? y : 0.f;
        o += y * out_w[j];
    }
    out[b] = 1.f / (1.f + expf(-o));
}

// ---------------------------------------------------------------------------
extern "C" void kernel_launch(void* const* d_in, const int* in_sizes, int n_in,
                              void* d_out, int out_size, void* d_ws, size_t ws_size,
                              hipStream_t stream) {
    const float* conts  = (const float*)d_in[0];
    const int*   cates  = (const int*)d_in[1];
    const float* emb    = (const float*)d_in[3];
    const float* sub_w0 = (const float*)d_in[4];
    const float* sub_b0 = (const float*)d_in[5];
    const float* sln0_w = (const float*)d_in[6];
    const float* sln0_b = (const float*)d_in[7];
    const float* sub_w1 = (const float*)d_in[8];
    const float* sub_b1 = (const float*)d_in[9];
    const float* sln1_w = (const float*)d_in[10];
    const float* sln1_b = (const float*)d_in[11];
    const float* fc0_w  = (const float*)d_in[12];
    const float* fc0_b  = (const float*)d_in[13];
    const float* ln0_w  = (const float*)d_in[14];
    const float* ln0_b  = (const float*)d_in[15];
    const float* fc1_w  = (const float*)d_in[16];
    const float* fc1_b  = (const float*)d_in[17];
    const float* ln1_w  = (const float*)d_in[18];
    const float* ln1_b  = (const float*)d_in[19];
    const float* out_w  = (const float*)d_in[20];
    const float* out_b  = (const float*)d_in[21];

    char* ws = (char*)d_ws;
    __bf16*   Xb  = (__bf16*)ws;                                 // BATCH*KX bf16
    unsigned* Wbp = (unsigned*)(ws + (size_t)BATCH * KX * 2);    // KSTEPS*7*256 u32
    float*    h0  = (float*)(ws + (size_t)BATCH * KX * 2
                                + (size_t)KSTEPS * NT_FC0 * 256 * 4);
    float* out = (float*)d_out;

    k_embed<<<BATCH, 256, 0, stream>>>(conts, cates, emb, Xb);
    int nw = KSTEPS * NT_FC0 * 256;
    k_w0cvt<<<(nw + 255) / 256, 256, 0, stream>>>(fc0_w, Wbp);
    dim3 gB(COMB, BATCH / 64);
    k_comb<<<gB, 128, 0, stream>>>(Xb, sub_w0, sub_b0, sln0_w, sln0_b,
                                   sub_w1, sub_b1, sln1_w, sln1_b);
    k_fc0<<<BATCH / 64, 128, 0, stream>>>(Xb, Wbp, fc0_b, ln0_w, ln0_b, h0);
    k_head<<<BATCH / 64, 64, 0, stream>>>(h0, fc1_w, fc1_b, ln1_w, ln1_b,
                                          out_w, out_b, out);
}
